// Attention_44744969290348
// MI455X (gfx1250) — matrix operand order repaired and checked by reading, not compiled, not run on hardware
//
#include <hip/hip_runtime.h>
#include <cstdint>
#include <cstddef>

// Problem constants (match reference)
#define BN 4
#define SN 4096
#define DN 256
#define NROW (BN * SN)   // 16384
#define EPSLN 1e-5f

typedef __attribute__((ext_vector_type(16))) _Float16 v16h;
typedef __attribute__((ext_vector_type(8)))  _Float16 v8h;
typedef __attribute__((ext_vector_type(8)))  float    v8f;
typedef __attribute__((ext_vector_type(4)))  int      v4i;

// ---------------------------------------------------------------------------
// tanh: gfx1250 has V_TANH_F32 (TRANS op). Use the builtin when present.
// ---------------------------------------------------------------------------
__device__ inline float fast_tanh(float x) {
#if __has_builtin(__builtin_amdgcn_tanhf)
    return __builtin_amdgcn_tanhf(x);
#else
    return tanhf(x);
#endif
}

// ---------------------------------------------------------------------------
// Async global->LDS copy (CDNA5 GLOBAL_LOAD_ASYNC_TO_LDS_B128, ASYNCcnt).
// Toolchain signature (from diagnostics): (v4i AS(1)*, v4i AS(3)*, imm, imm).
// Falls back to load/store staging when the builtin is unavailable.
// ---------------------------------------------------------------------------
#if __has_builtin(__builtin_amdgcn_global_load_async_to_lds_b128)
#define USE_ASYNC_LDS 1
typedef __attribute__((address_space(1))) v4i* gv4i_p;
typedef __attribute__((address_space(3))) v4i* lv4i_p;
#endif

__device__ inline void copy16_g2l(const _Float16* g, _Float16* l) {
#ifdef USE_ASYNC_LDS
    // AS(3) LDS pointer = low 32 bits of the generic LDS aperture address
    __builtin_amdgcn_global_load_async_to_lds_b128(
        (gv4i_p)(uintptr_t)g,
        (lv4i_p)(uintptr_t)(uint32_t)(uintptr_t)l,
        0, 0);
#else
    *(v8h*)l = *(const v8h*)g;
#endif
}

__device__ inline void async_copies_wait() {
#ifdef USE_ASYNC_LDS
#if __has_builtin(__builtin_amdgcn_s_wait_asynccnt)
    __builtin_amdgcn_s_wait_asynccnt(0);
#else
    asm volatile("s_wait_asynccnt 0x0" ::: "memory");
#endif
#endif
}

// ---------------------------------------------------------------------------
// WMMA fragment loaders (wave32, v_wmma_f32_16x16x32_f16, contiguous k order used for both A and B)
// A (16x32 f16, M x K): lane m = lane&15, khalf = (lane>>4)*16
//   elements 0..15 <- K = khalf .. khalf+15
// B (32x16 f16, K x N): lane n = lane&15, kbase = (lane>>4)*16
//   elements 0..15 <- K = kbase .. kbase+15
// C/D (16x16 f32): lane n = lane&15, row m = (lane>>4)*8 + r
// ---------------------------------------------------------------------------

__device__ inline v16h load_a_f16(const _Float16* tile, int ld, int lane) {
    int m  = lane & 15;
    int kh = (lane >> 4) * 16;
    const _Float16* p = tile + (size_t)m * ld + kh;
    v8h lo = *(const v8h*)(p);
    v8h hi = *(const v8h*)(p + 8);
    v16h f;
#pragma unroll
    for (int i = 0; i < 8; ++i) { f[i] = lo[i]; f[i + 8] = hi[i]; }
    return f;
}

// B fragment from a 16B-aligned (not necessarily 32B) location (LDS rows)
__device__ inline v16h load_b_pair(const _Float16* p) {
    v8h lo = *(const v8h*)(p);
    v8h hi = *(const v8h*)(p + 8);
    v16h f;
#pragma unroll
    for (int i = 0; i < 8; ++i) { f[i] = lo[i]; f[i + 8] = hi[i]; }
    return f;
}

__device__ inline v16h load_a_f32(const float* tile, int ld, int lane) {
    int m  = lane & 15;
    int kh = (lane >> 4) * 16;
    const float* p = tile + (size_t)m * ld + kh;
    v8f lo = *(const v8f*)(p);
    v8f hi = *(const v8f*)(p + 8);
    v16h f;
#pragma unroll
    for (int i = 0; i < 8; ++i) { f[i] = (_Float16)lo[i]; f[i + 8] = (_Float16)hi[i]; }
    return f;
}

__device__ inline v16h load_b_f32(const float* tile, int ld, int lane) {
    int n  = lane & 15;
    int kb = (lane >> 4) * 16;
    const float* p = tile + (size_t)n * ld + kb;
    v8f lo = *(const v8f*)(p);
    v8f hi = *(const v8f*)(p + 8);
    v16h f;
#pragma unroll
    for (int i = 0; i < 8; ++i) { f[i] = (_Float16)lo[i]; f[i + 8] = (_Float16)hi[i]; }
    return f;
}

// ---------------------------------------------------------------------------
// Kernel 0: zero the score accumulator s[2][B][S]
// ---------------------------------------------------------------------------
__global__ void zero_scores_kernel(float* s, int n) {
    int i = blockIdx.x * blockDim.x + threadIdx.x;
    if (i < n) s[i] = 0.0f;
}

// ---------------------------------------------------------------------------
// Kernel 1: fused projections  relu(x @ W.T + b)
//   grid.x = NROW/16 (row strip), grid.y = 6 : (x1|x2) x (Wq|Wk|Wv)
//   q/k outputs stored f16 (feed WMMA attention), v outputs stored f32.
// ---------------------------------------------------------------------------
__global__ __launch_bounds__(32)
void proj_kernel(const float* __restrict__ x1, const float* __restrict__ x2,
                 const float* __restrict__ Wq, const float* __restrict__ bq,
                 const float* __restrict__ Wk, const float* __restrict__ bk,
                 const float* __restrict__ Wv, const float* __restrict__ bv,
                 _Float16* __restrict__ q1h, _Float16* __restrict__ q2h,
                 _Float16* __restrict__ k1h, _Float16* __restrict__ k2h,
                 float* __restrict__ v1f, float* __restrict__ v2f) {
    const int lane = threadIdx.x;
    const int g    = blockIdx.y;
    const int xsel = g / 3;        // 0: x1, 1: x2
    const int wsel = g % 3;        // 0: Wq, 1: Wk, 2: Wv

    const float* X    = xsel ? x2 : x1;
    const float* W    = (wsel == 0) ? Wq : ((wsel == 1) ? Wk : Wv);
    const float* bias = (wsel == 0) ? bq : ((wsel == 1) ? bk : bv);

    _Float16* oh = nullptr;
    float*    of = nullptr;
    if (wsel == 0)      oh = xsel ? q2h : q1h;
    else if (wsel == 1) oh = xsel ? k2h : k1h;
    else                of = xsel ? v2f : v1f;

    const int it = blockIdx.x;                  // row tile
    v16h af[8];
#pragma unroll
    for (int kk = 0; kk < 8; ++kk)
        af[kk] = load_a_f32(X + (size_t)it * 16 * DN + kk * 32, DN, lane);

    const int n_in_tile = lane & 15;
    const int mbase     = it * 16 + (lane >> 4) * 8;

    for (int nt = 0; nt < 16; ++nt) {
        v8f c = {0.f, 0.f, 0.f, 0.f, 0.f, 0.f, 0.f, 0.f};
#pragma unroll
        for (int kk = 0; kk < 8; ++kk) {
            v16h bf = load_b_f32(W + (size_t)nt * 16 * DN + kk * 32, DN, lane);
            c = __builtin_amdgcn_wmma_f32_16x16x32_f16(
                    false, af[kk], false, bf, (short)0, c, false, false);
        }
        const int   ncol = nt * 16 + n_in_tile;
        const float bb   = bias[ncol];
#pragma unroll
        for (int r = 0; r < 8; ++r) {
            float val = c[r] + bb;
            val = val > 0.0f ? val : 0.0f;
            size_t idx = (size_t)(mbase + r) * DN + ncol;
            if (oh) oh[idx] = (_Float16)val;
            else    of[idx] = val;
        }
    }
}

// ---------------------------------------------------------------------------
// Kernel 2: s[side][b][i] += sum_j tanh( K[b,i,:] . Q[b,j,:] )
//   side 0: K=k1, Q=q2 ; side 1: K=k2, Q=q1
// Block = 256 threads (8 waves). Each wave owns TWO 16-row A strips (32 rows,
// 128 VGPRs) held for the whole j loop -> workgroup covers 256 K rows.
// Q is staged through double-buffered LDS in 64-row blocks via async
// global->LDS copies (rows padded to 264 halves to spread DS banks); each B
// fragment read from LDS feeds 2 WMMAs.
//   grid = (SN/256, 2*B, JC) ; JC splits the j range, partials atomicAdd'ed.
// ---------------------------------------------------------------------------
#define JC     2
#define JBROWS 64
#define LDSLD  264           // padded row stride in halves (528 B)

__global__ __launch_bounds__(256)
void scores_kernel(const _Float16* __restrict__ k1h, const _Float16* __restrict__ k2h,
                   const _Float16* __restrict__ q1h, const _Float16* __restrict__ q2h,
                   float* __restrict__ s) {
    __shared__ __align__(32) _Float16 lbuf[2][JBROWS * LDSLD];   // 2 x ~33 KB

    const int tid  = threadIdx.x;
    const int lane = tid & 31;
    const int wv   = tid >> 5;          // wave id 0..7
    const int by   = blockIdx.y;
    const int side = by >> 2;
    const int b    = by & 3;

    const _Float16* A  = ((side == 0) ? k1h : k2h) + (size_t)b * SN * DN;
    const _Float16* Qb = ((side == 0) ? q2h : q1h) + (size_t)b * SN * DN;
    float* srow = s + ((size_t)side * BN + b) * SN;

    // two A strips per wave: rows [it0*16, it0*16+32)
    const int it0 = blockIdx.x * 16 + wv * 2;
    v16h af0[8], af1[8];
#pragma unroll
    for (int kk = 0; kk < 8; ++kk) {
        af0[kk] = load_a_f16(A + (size_t)it0 * 16 * DN + kk * 32, DN, lane);
        af1[kk] = load_a_f16(A + (size_t)(it0 + 1) * 16 * DN + kk * 32, DN, lane);
    }

    float rs0[8], rs1[8];
#pragma unroll
    for (int r = 0; r < 8; ++r) { rs0[r] = 0.0f; rs1[r] = 0.0f; }

    const int jrow0 = blockIdx.z * (SN / JC);            // chunk start row
    const int nblk  = (SN / JC) / JBROWS;                // blocks per chunk

    // stage(buf, jbase): 64 rows x 512 B, 16 B chunks, 8 per thread
    auto stage = [&](int buf, int jbase) {
        const _Float16* src = Qb + (size_t)jbase * DN;
#pragma unroll
        for (int t = 0; t < (JBROWS * 32) / 256; ++t) {
            int i   = tid + t * 256;
            int row = i >> 5;
            int seg = i & 31;
            copy16_g2l(src + (size_t)row * DN + seg * 8,
                       &lbuf[buf][row * LDSLD + seg * 8]);
        }
#ifndef USE_ASYNC_LDS
        // touch the block after next so it is resident in cache
        if (jbase + JBROWS < jrow0 + SN / JC) {
            const _Float16* nxt = src + (size_t)JBROWS * DN;
#pragma unroll
            for (int t = 0; t < (JBROWS * 32) / 256; ++t) {
                int i = tid + t * 256;
                __builtin_prefetch(nxt + (size_t)(i >> 5) * DN + (i & 31) * 8, 0, 0);
            }
        }
#endif
    };

    const int n  = lane & 15;
    const int kb = (lane >> 4) * 16;

    int cur = 0;
    stage(cur, jrow0);
    for (int blk = 0; blk < nblk; ++blk) {
        async_copies_wait();      // my async fills are done (no-op in fallback)
        __syncthreads();          // staged data visible; prev buffer consumed

        // overlap: start filling the other buffer while we compute this one
        if (blk + 1 < nblk) stage(cur ^ 1, jrow0 + (blk + 1) * JBROWS);

        // --- compute: 4 j-tiles per block, B frags from LDS, 2 WMMAs each ---
#pragma unroll
        for (int jt = 0; jt < JBROWS / 16; ++jt) {
            v8f c0 = {0.f, 0.f, 0.f, 0.f, 0.f, 0.f, 0.f, 0.f};
            v8f c1 = {0.f, 0.f, 0.f, 0.f, 0.f, 0.f, 0.f, 0.f};
            const _Float16* brow = &lbuf[cur][(jt * 16 + n) * LDSLD + kb];
#pragma unroll
            for (int kk = 0; kk < 8; ++kk) {
                v16h bf = load_b_pair(brow + kk * 32);
                c0 = __builtin_amdgcn_wmma_f32_16x16x32_f16(
                        false, af0[kk], false, bf, (short)0, c0, false, false);
                c1 = __builtin_amdgcn_wmma_f32_16x16x32_f16(
                        false, af1[kk], false, bf, (short)0, c1, false, false);
            }
#pragma unroll
            for (int r = 0; r < 8; ++r) {
                rs0[r] += fast_tanh(c0[r]);
                rs1[r] += fast_tanh(c1[r]);
            }
        }
        cur ^= 1;
    }

    // Reduce across the 16 N-lanes of each half-wave (wave32, xor bits 0..3)
#pragma unroll
    for (int off = 1; off < 16; off <<= 1) {
#pragma unroll
        for (int r = 0; r < 8; ++r) {
            rs0[r] += __shfl_xor(rs0[r], off, 32);
            rs1[r] += __shfl_xor(rs1[r], off, 32);
        }
    }
    if ((lane & 15) == 0) {
        const int mb0 = it0 * 16 + (lane >> 4) * 8;
        const int mb1 = (it0 + 1) * 16 + (lane >> 4) * 8;
#pragma unroll
        for (int r = 0; r < 8; ++r) {
            atomicAdd(&srow[mb0 + r], rs0[r]);
            atomicAdd(&srow[mb1 + r], rs1[r]);
        }
    }
}

// ---------------------------------------------------------------------------
// Kernel 3: masked softmax over S, vec = a.V, y = mean(x)+vec, layernorm.
//   grid = (B, 2) ; block = 256 (one thread per feature dim)
// ---------------------------------------------------------------------------
__global__ __launch_bounds__(256)
void finalize_kernel(const float* __restrict__ x1, const float* __restrict__ x2,
                     const unsigned char* __restrict__ m1, const unsigned char* __restrict__ m2,
                     const float* __restrict__ v1f, const float* __restrict__ v2f,
                     const float* __restrict__ s,
                     const float* __restrict__ gamma, const float* __restrict__ beta,
                     float* __restrict__ out) {
    __shared__ float wts[SN];
    __shared__ float red[256];

    const int tid  = threadIdx.x;
    const int b    = blockIdx.x;
    const int side = blockIdx.y;

    const float* svec        = s + ((size_t)side * BN + b) * SN;
    const unsigned char* msk = ((side == 0) ? m1 : m2) + (size_t)b * SN;
    const float* V           = ((side == 0) ? v1f : v2f) + (size_t)b * SN * DN;
    const float* X           = ((side == 0) ? x1  : x2 ) + (size_t)b * SN * DN;

    // row max of masked scores (mask==true -> -inf)
    float lm = -INFINITY;
    for (int i = tid; i < SN; i += 256) {
        float v = msk[i] ? -INFINITY : svec[i];
        lm = fmaxf(lm, v);
    }
    red[tid] = lm; __syncthreads();
    for (int st = 128; st > 0; st >>= 1) {
        if (tid < st) red[tid] = fmaxf(red[tid], red[tid + st]);
        __syncthreads();
    }
    const float mx = red[0];
    __syncthreads();

    // exp + sum
    float lz = 0.0f;
    for (int i = tid; i < SN; i += 256) {
        float v = msk[i] ? -INFINITY : svec[i];
        float e = expf(v - mx);     // expf(-inf) == 0
        wts[i] = e;
        lz += e;
    }
    red[tid] = lz; __syncthreads();
    for (int st = 128; st > 0; st >>= 1) {
        if (tid < st) red[tid] += red[tid + st];
        __syncthreads();
    }
    const float inv = 1.0f / red[0];
    __syncthreads();

    // per-dim fused mean(x) + a.V (coalesced column walk)
    const int d = tid;
    float accx = 0.0f, accv = 0.0f;
    for (int i = 0; i < SN; ++i) {
        accx += X[(size_t)i * DN + d];
        accv += wts[i] * V[(size_t)i * DN + d];
    }
    float y = accx * (1.0f / (float)SN) + accv * inv;

    // layernorm over 256 dims
    red[tid] = y; __syncthreads();
    for (int st = 128; st > 0; st >>= 1) {
        if (tid < st) red[tid] += red[tid + st];
        __syncthreads();
    }
    const float mu = red[0] * (1.0f / (float)DN);
    __syncthreads();
    float dy = y - mu;
    red[tid] = dy * dy; __syncthreads();
    for (int st = 128; st > 0; st >>= 1) {
        if (tid < st) red[tid] += red[tid + st];
        __syncthreads();
    }
    const float var = red[0] * (1.0f / (float)DN);

    out[((size_t)side * BN + b) * DN + d] =
        dy * rsqrtf(var + EPSLN) * gamma[d] + beta[d];
}

// ---------------------------------------------------------------------------
// Host launcher
// ---------------------------------------------------------------------------
extern "C" void kernel_launch(void* const* d_in, const int* in_sizes, int n_in,
                              void* d_out, int out_size, void* d_ws, size_t ws_size,
                              hipStream_t stream) {
    const float* x1 = (const float*)d_in[0];
    const float* x2 = (const float*)d_in[1];
    const unsigned char* m1 = (const unsigned char*)d_in[2];   // jnp bool -> 1 byte
    const unsigned char* m2 = (const unsigned char*)d_in[3];
    const float* Wq = (const float*)d_in[4];
    const float* bq = (const float*)d_in[5];
    const float* Wk = (const float*)d_in[6];
    const float* bk = (const float*)d_in[7];
    const float* Wv = (const float*)d_in[8];
    const float* bv = (const float*)d_in[9];
    const float* gamma = (const float*)d_in[10];
    const float* beta  = (const float*)d_in[11];
    float* out = (float*)d_out;

    // workspace layout (all offsets 256B aligned)
    char* ws = (char*)d_ws;
    const size_t halfMat = (size_t)NROW * DN * sizeof(_Float16);   // 8 MB
    const size_t fltMat  = (size_t)NROW * DN * sizeof(float);      // 16 MB
    _Float16* q1h = (_Float16*)(ws);
    _Float16* q2h = (_Float16*)(ws + halfMat);
    _Float16* k1h = (_Float16*)(ws + 2 * halfMat);
    _Float16* k2h = (_Float16*)(ws + 3 * halfMat);
    float*    v1f = (float*)   (ws + 4 * halfMat);
    float*    v2f = (float*)   (ws + 4 * halfMat + fltMat);
    float*    s   = (float*)   (ws + 4 * halfMat + 2 * fltMat);    // [2][B][S]

    // zero score accumulator
    {
        int n = 2 * BN * SN;
        zero_scores_kernel<<<(n + 255) / 256, 256, 0, stream>>>(s, n);
    }
    // projections (6 GEMMs fused into one grid)
    proj_kernel<<<dim3(NROW / 16, 6), 32, 0, stream>>>(
        x1, x2, Wq, bq, Wk, bk, Wv, bv, q1h, q2h, k1h, k2h, v1f, v2f);
    // fused K.Q^T -> tanh -> rowsum (async double-buffered LDS staging)
    scores_kernel<<<dim3(SN / 256, 2 * BN, JC), 256, 0, stream>>>(
        k1h, k2h, q1h, q2h, s);
    // softmax + a.V + mean + layernorm
    finalize_kernel<<<dim3(BN, 2), 256, 0, stream>>>(
        x1, x2, m1, m2, v1f, v2f, s, gamma, beta, out);
}
